// ScaledDotProductAttention_84327387889697
// MI455X (gfx1250) — matrix operand order repaired and checked by reading, not compiled, not run on hardware
//
#include <hip/hip_runtime.h>
#include <hip/hip_bf16.h>
#include <math.h>
#include <stdint.h>

typedef __attribute__((ext_vector_type(16))) _Float16 v16h;
typedef __attribute__((ext_vector_type(8)))  float    v8f;
typedef __attribute__((ext_vector_type(2)))  _Float16 h2;

#define N_B 2
#define H_H 4
#define LQ 1024
#define LK 1024
#define DH 64
#define RR 128
#define NP (2*RR+1)   /* 257 */
#define NP_PAD 320    /* bias rows padded so stores need no bounds check */

// K-offset of the f16 pair held in A-matrix VGPR i for lane-half g
// (contiguous order: k = 16*g + 2*i, matching the B fragments)
__device__ __forceinline__ int kbase16(int i, int g) {
  return g * 16 + i * 2;
}

__device__ __forceinline__ v8f wmma16(v16h a, v16h b, v8f c) {
  // D = A(16x32 f16) * B(32x16 f16) + C(16x16 f32)
  return __builtin_amdgcn_wmma_f32_16x16x32_f16(false, a, false, b,
                                                (short)0, c, false, false);
}

// ---------------------------------------------------------------------------
// Kernel 0: per-batch scaling = sqrt(#valid keys)
// ---------------------------------------------------------------------------
__global__ void scaling_kernel(const unsigned char* __restrict__ pad,
                               float* __restrict__ wsScale) {
  __shared__ int red[256];
  int n = blockIdx.x;
  int t = threadIdx.x;
  int cnt = 0;
  for (int i = t; i < LK; i += 256) cnt += (pad[n * LK + i] == 0) ? 1 : 0;
  red[t] = cnt;
  __syncthreads();
  for (int s = 128; s > 0; s >>= 1) {
    if (t < s) red[t] += red[t + s];
    __syncthreads();
  }
  if (t == 0) wsScale[n] = sqrtf((float)red[0]);
}

// ---------------------------------------------------------------------------
// Kernel 1: bias[nh][p][k] = dot(rpe_weight[p, h*64 : h*64+64], K[nh,k,:])
// GEMM M=257(padded to 320) x N=1024 x K=64 per head, via WMMA f16.
// block = 128 threads (4 waves); wave handles 16 p-rows x 64 key-cols.
// ---------------------------------------------------------------------------
__global__ __launch_bounds__(128)
void rpe_bias_kernel(const float* __restrict__ kmat,
                     const float* __restrict__ rpe,
                     float* __restrict__ bias) {
  int lane = threadIdx.x & 31;
  int wave = threadIdx.x >> 5;
  int g  = lane >> 4;
  int ln = lane & 15;
  int k0 = blockIdx.x * 64;
  int pbase = (blockIdx.y * 4 + wave) * 16;
  int nh = blockIdx.z;
  int h = nh & (H_H - 1);
  const float* kp = kmat + (size_t)nh * LK * DH;

  // A fragments: rpe rows (clamped to 2R for the overhang rows)
  int pr = pbase + ln; if (pr > 2 * RR) pr = 2 * RR;
  const float* rrow = rpe + (size_t)pr * (H_H * DH) + h * DH;
  v16h a0, a1;
#pragma unroll
  for (int i = 0; i < 8; ++i) {
    int kb = kbase16(i, g);
    float2 x0 = *(const float2*)(rrow + kb);
    float2 x1 = *(const float2*)(rrow + 32 + kb);
    a0[2*i] = (_Float16)x0.x; a0[2*i+1] = (_Float16)x0.y;
    a1[2*i] = (_Float16)x1.x; a1[2*i+1] = (_Float16)x1.y;
  }

#pragma unroll
  for (int t = 0; t < 4; ++t) {
    int key = k0 + t * 16 + ln;
    const float* krow = kp + (size_t)key * DH;
    // B fragments: B[d][key] = K[key][d]; VGPR i holds d rows g*16+2i, +1
    v16h b0, b1;
#pragma unroll
    for (int i = 0; i < 8; ++i) {
      int d = g * 16 + 2 * i;
      float2 y0 = *(const float2*)(krow + d);
      float2 y1 = *(const float2*)(krow + 32 + d);
      b0[2*i] = (_Float16)y0.x; b0[2*i+1] = (_Float16)y0.y;
      b1[2*i] = (_Float16)y1.x; b1[2*i+1] = (_Float16)y1.y;
    }
    v8f c = {};
    c = wmma16(a0, b0, c);
    c = wmma16(a1, b1, c);
    // C layout: row = vgpr + 8*half; rows padded to NP_PAD -> unconditional store
#pragma unroll
    for (int j = 0; j < 8; ++j) {
      int p = pbase + j + 8 * g;
      bias[((size_t)nh * NP_PAD + p) * LK + key] = c[j];
    }
  }
}

// ---------------------------------------------------------------------------
// Kernel 2: flash attention. block = 128 (4 waves); wave owns 16 q-rows.
// grid = (Lq/64, N*H). V tile staged via async global->LDS scatter (transpose).
// ---------------------------------------------------------------------------
__global__ __launch_bounds__(128)
void attn_kernel(const float* __restrict__ q, const float* __restrict__ kmat,
                 const float* __restrict__ v, const unsigned char* __restrict__ pad,
                 const unsigned char* __restrict__ mf_p, const int* __restrict__ moff_p,
                 const float* __restrict__ wsScale, const float* __restrict__ bias,
                 float* __restrict__ out) {
  __shared__ float    vtf[DH][34];     // V^T tile f32: [d][key] (stride 34: conflict-free)
  __shared__ _Float16 ps[4][16][34];   // per-wave P tile: [row][key]

  int lane = threadIdx.x & 31;
  int wave = threadIdx.x >> 5;
  int g  = lane >> 4;
  int ln = lane & 15;
  int qt = blockIdx.x * 64;
  int qbase = qt + wave * 16;
  int nh = blockIdx.y;
  int n  = nh >> 2;  // /H
  const float* qp = q    + (size_t)nh * LQ * DH;
  const float* kp = kmat + (size_t)nh * LK * DH;
  const float* vp = v    + (size_t)nh * LK * DH;
  const float* bp = bias + (size_t)nh * NP_PAD * LK;
  int maskF = (int)mf_p[0];
  int moff  = moff_p[0];
  float inv_s = 1.0f / wsScale[n];

  // Q A-fragments (resident for the whole kernel)
  v16h a0, a1;
  {
    const float* qrow = qp + (size_t)(qbase + ln) * DH;
#pragma unroll
    for (int i = 0; i < 8; ++i) {
      int kb = kbase16(i, g);
      float2 x0 = *(const float2*)(qrow + kb);
      float2 x1 = *(const float2*)(qrow + 32 + kb);
      a0[2*i] = (_Float16)x0.x; a0[2*i+1] = (_Float16)x0.y;
      a1[2*i] = (_Float16)x1.x; a1[2*i+1] = (_Float16)x1.y;
    }
  }

  v8f o[4] = {};
  float m_run[8], l_run[8];
#pragma unroll
  for (int j = 0; j < 8; ++j) { m_run[j] = -1e30f; l_run[j] = 0.0f; }

  int kend = maskF ? (LK < qt + 64 + moff ? LK : qt + 64 + moff) : LK;

  for (int k0 = 0; k0 < kend; k0 += 32) {
    __syncthreads();  // protect vtf from previous-iteration readers

    // Stage V^T via CDNA5 async global->LDS (B32 scatter does the transpose):
    // global side coalesced over d, LDS side strided by row 34*4B.
#pragma unroll
    for (int rep = 0; rep < (32 * DH) / 128; ++rep) {
      int idx = rep * 128 + threadIdx.x;
      int key = idx >> 6, d = idx & 63;
      unsigned ldsoff = (unsigned)(uintptr_t)&vtf[d][key];
      const float* gaddr = vp + (size_t)(k0 + key) * DH + d;
      asm volatile("global_load_async_to_lds_b32 %0, %1, off"
                   :: "v"(ldsoff), "v"(gaddr) : "memory");
    }
    asm volatile("s_wait_asynccnt 0x0" ::: "memory");
    __syncthreads();

    // ---- scores for two 16-col tiles ----
    float sv[2][8];
#pragma unroll
    for (int t = 0; t < 2; ++t) {
      int col = k0 + t * 16 + ln;
      const float* krow = kp + (size_t)col * DH;
      v16h b0, b1;
#pragma unroll
      for (int i = 0; i < 8; ++i) {
        int d = g * 16 + 2 * i;
        float2 y0 = *(const float2*)(krow + d);
        float2 y1 = *(const float2*)(krow + 32 + d);
        b0[2*i] = (_Float16)y0.x; b0[2*i+1] = (_Float16)y0.y;
        b1[2*i] = (_Float16)y1.x; b1[2*i+1] = (_Float16)y1.y;
      }
      v8f s = {};
      s = wmma16(a0, b0, s);
      s = wmma16(a1, b1, s);
      int padded = pad[(size_t)n * LK + col];
#pragma unroll
      for (int j = 0; j < 8; ++j) {
        int row = qbase + j + 8 * g;
        int p = RR + col - row - moff;
        p = p < 0 ? 0 : (p > 2 * RR ? 2 * RR : p);
        float val = (s[j] + bp[(size_t)p * LK + col]) * inv_s;
        if ((maskF && col > row + moff) || padded) val = -1e30f;
        sv[t][j] = val;
      }
    }

    // ---- online softmax (row reductions over 16-lane halves) ----
    float alpha[8];
#pragma unroll
    for (int j = 0; j < 8; ++j) {
      float mloc = fmaxf(sv[0][j], sv[1][j]);
#pragma unroll
      for (int off = 1; off <= 8; off <<= 1)
        mloc = fmaxf(mloc, __shfl_xor(mloc, off, 32));
      float mnew = fmaxf(m_run[j], mloc);
      alpha[j] = __expf(m_run[j] - mnew);
      float p0 = (sv[0][j] <= -1e29f) ? 0.0f : __expf(sv[0][j] - mnew);
      float p1 = (sv[1][j] <= -1e29f) ? 0.0f : __expf(sv[1][j] - mnew);
      float lloc = p0 + p1;
#pragma unroll
      for (int off = 1; off <= 8; off <<= 1)
        lloc += __shfl_xor(lloc, off, 32);
      l_run[j] = l_run[j] * alpha[j] + lloc;
      m_run[j] = mnew;
      ps[wave][j + 8 * g][ln]      = (_Float16)p0;   // C layout -> LDS [row][col]
      ps[wave][j + 8 * g][16 + ln] = (_Float16)p1;
    }
#pragma unroll
    for (int dt = 0; dt < 4; ++dt)
#pragma unroll
      for (int j = 0; j < 8; ++j) o[dt][j] *= alpha[j];

    __syncthreads();  // P LDS store -> A-layout reload (cross-lane within wave)

    // reload P in A-matrix layout
    v16h pa;
#pragma unroll
    for (int i = 0; i < 8; ++i) {
      int kb = kbase16(i, g);
      h2 pr = *(const h2*)&ps[wave][ln][kb];
      pa[2*i] = pr.x; pa[2*i+1] = pr.y;
    }

    // O += P * V  (B frags from transposed f32 V tile: pair = 2 consecutive keys)
#pragma unroll
    for (int dt = 0; dt < 4; ++dt) {
      int d = dt * 16 + ln;
      v16h vb;
#pragma unroll
      for (int i = 0; i < 8; ++i) {
        int kk = g * 16 + 2 * i;
        float2 pr = *(const float2*)&vtf[d][kk];
        vb[2*i] = (_Float16)pr.x; vb[2*i+1] = (_Float16)pr.y;
      }
      o[dt] = wmma16(pa, vb, o[dt]);
    }
  }

  // ---- epilogue: out = O / l ----
#pragma unroll
  for (int j = 0; j < 8; ++j) {
    int row = qbase + j + 8 * g;
    float il = (l_run[j] > 0.0f) ? 1.0f / l_run[j] : 0.0f;
    float* orow = out + ((size_t)nh * LQ + row) * DH;
#pragma unroll
    for (int dt = 0; dt < 4; ++dt)
      orow[dt * 16 + ln] = o[dt][j] * il;
  }
}

// ---------------------------------------------------------------------------
extern "C" void kernel_launch(void* const* d_in, const int* in_sizes, int n_in,
                              void* d_out, int out_size, void* d_ws, size_t ws_size,
                              hipStream_t stream) {
  const float* q   = (const float*)d_in[0];
  const float* k   = (const float*)d_in[1];
  const float* v   = (const float*)d_in[2];
  const float* rpe = (const float*)d_in[3];
  const unsigned char* pad = (const unsigned char*)d_in[4];
  const unsigned char* mf  = (const unsigned char*)d_in[5];
  const int* moff          = (const int*)d_in[6];
  float* out = (float*)d_out;

  // workspace layout: [0,256) per-batch scaling; then bias[N*H][320][Lk] f32
  float* wsScale = (float*)d_ws;
  float* bias    = (float*)((char*)d_ws + 256);

  scaling_kernel<<<N_B, 256, 0, stream>>>(pad, wsScale);
  rpe_bias_kernel<<<dim3(LK / 64, 5, N_B * H_H), 128, 0, stream>>>(k, rpe, bias);
  attn_kernel<<<dim3(LQ / 64, N_B * H_H), 128, 0, stream>>>(
      q, k, v, pad, mf, moff, wsScale, bias, out);
}
